// CharLSTM_77773267796710
// MI455X (gfx1250) — matrix operand order repaired and checked by reading, not compiled, not run on hardware
//
#include <hip/hip_runtime.h>

// ---------------------------------------------------------------------------
// CharLSTM on MI455X (gfx1250), bf16 WMMA.
// Workgroup = 16-row batch tile, wave = 16-col hidden tile: the LSTM
// recurrence never mixes batch rows, so all h-state dependencies are
// intra-workgroup (h in LDS, sync via s_barrier); the 32 workgroups are
// fully independent. Weights stream per-step from the L2-resident packed
// buffers (an explicit memory clobber stops LICM from hoisting+spilling them).
//   gates0 = G0[x[b,t]] + h0 @ Whh0^T            (G0 folds E/Wih0/biases)
//   gates1 = b1 + h0_t @ Wih1^T + h1_{t-1} @ Whh1^T
//   out    = h1_{T-1} @ Wfc^T + bfc
// c-state lives in fp32 VGPRs for the whole sequence.
// ---------------------------------------------------------------------------

#define BATCH 512
#define SEQ   512
#define HID   256
#define G4    1024           // 4*HID
#define VOCAB 78
#define NWG   32             // one per 16-row batch tile
#define NTHR  512            // 16 wave32 waves: one per 16-col hidden tile
#define LSTR  264            // LDS h-tile row stride in bf16 (528B: bank-skewed, 16B aligned)

typedef __attribute__((ext_vector_type(16))) __bf16 v16bf;
typedef __attribute__((ext_vector_type(8)))  __bf16 v8bf;
typedef __attribute__((ext_vector_type(8)))  float  v8f;

// --- workspace layout (bytes) ---
#define G0_OFF    0u                      // 78*1024 f32 (319488 -> pad)
#define B1_OFF    327680u                 // 1024 f32
#define W0P_OFF   331776u                 // Whh0 packed bf16 [1024*256]
#define W1IP_OFF  856064u                 // Wih1 packed bf16
#define W1HP_OFF  1380352u                // Whh1 packed bf16
#define WS_NEED   1904640u

__device__ __forceinline__ float bf2f(__bf16 x) {
  unsigned short s = __builtin_bit_cast(unsigned short, x);
  unsigned u = ((unsigned)s) << 16;
  return __builtin_bit_cast(float, u);
}
__device__ __forceinline__ __bf16 f2bf(float f) {   // round-to-nearest-even
  unsigned u = __builtin_bit_cast(unsigned, f);
  unsigned r = u + 0x7FFFu + ((u >> 16) & 1u);
  return __builtin_bit_cast(__bf16, (unsigned short)(r >> 16));
}
__device__ __forceinline__ float sigm(float x) { return 1.0f / (1.0f + __expf(-x)); }

// ---------------------------------------------------------------------------
// Prep 1: G0 gate table (embedding x Wih0 + both layer-0 biases; padding_idx=0
// => E row 0 treated as zero) and combined layer-1 bias b1.
// ---------------------------------------------------------------------------
__global__ __launch_bounds__(256) void prep_tables(
    const float* __restrict__ E, const float* __restrict__ Wih0,
    const float* __restrict__ bih0, const float* __restrict__ bhh0,
    const float* __restrict__ bih1, const float* __restrict__ bhh1,
    float* __restrict__ G0, float* __restrict__ b1) {
  int i = blockIdx.x * 256 + threadIdx.x;
  if (i < VOCAB * G4) {
    int v = i >> 10, g = i & 1023;
    float s = bih0[g] + bhh0[g];
    if (v != 0) {
#pragma unroll
      for (int k = 0; k < 8; ++k) s += E[v * 8 + k] * Wih0[g * 8 + k];
    }
    G0[i] = s;
  } else if (i < VOCAB * G4 + G4) {
    int g = i - VOCAB * G4;
    b1[g] = bih1[g] + bhh1[g];
  }
}

// ---------------------------------------------------------------------------
// Prep 2: pack W[1024][256] f32 -> bf16 WMMA-B fragment order.
// Fragment (nt,kb): element (L,e) = W[nt*16 + L%16][kb*32 + 16*(e>=8) + 8*(L>=16) + e%8]
// per the CDNA5 32x16 bf16 B-matrix lane layout.
// ---------------------------------------------------------------------------
__global__ __launch_bounds__(256) void prep_pack(
    const float* __restrict__ Whh0, const float* __restrict__ Wih1,
    const float* __restrict__ Whh1,
    __bf16* __restrict__ W0p, __bf16* __restrict__ W1ip, __bf16* __restrict__ W1hp) {
  int i = blockIdx.x * 256 + threadIdx.x;
  if (i >= 3 * G4 * HID) return;
  int m = i >> 18;
  int r = i & (G4 * HID - 1);
  int f = r >> 9, t = r & 511;
  int L = t >> 4, e = t & 15;
  int nt = f >> 3, kb = f & 7;
  int g = (nt << 4) + (L & 15);
  int k = (kb << 5) + ((e >> 3) << 4) + ((L >> 4) << 3) + (e & 7);
  const float* src = (m == 0) ? Whh0 : (m == 1 ? Wih1 : Whh1);
  __bf16* dst = (m == 0) ? W0p : (m == 1 ? W1ip : W1hp);
  dst[r] = f2bf(src[g * HID + k]);
}

// Load a 16x32 bf16 A-fragment (CDNA5 lane layout) from an LDS h tile
// ([16][LSTR] bf16): rows = local batch rows, K block kb.
__device__ __forceinline__ v16bf load_afrag_lds(const __bf16* h, int kb,
                                                int lrow, int lhi) {
  int base = lrow * LSTR + (kb << 5) + (lhi << 3);
  v8bf lo = *(const v8bf*)(h + base);
  v8bf hi = *(const v8bf*)(h + base + 16);
  return __builtin_shufflevector(lo, hi, 0, 1, 2, 3, 4, 5, 6, 7,
                                 8, 9, 10, 11, 12, 13, 14, 15);
}

// ---------------------------------------------------------------------------
// Main kernel: 32 independent WGs x 512 threads (16 waves).
// WG = batch rows [16*blk, 16*blk+16); wave wv = hidden tile jt = wv.
// Per step: layer0 (32 wmma/wave), s_barrier, layer1 (64 wmma/wave), s_barrier.
// ---------------------------------------------------------------------------
__global__ __launch_bounds__(NTHR, 1) void lstm_batchtile(
    const int* __restrict__ x,
    const float* __restrict__ G0, const float* __restrict__ b1,
    const __bf16* __restrict__ W0p, const __bf16* __restrict__ W1ip,
    const __bf16* __restrict__ W1hp,
    const float* __restrict__ Wfc, const float* __restrict__ bfc,
    float* __restrict__ out) {
  // 4 h tiles: [0..1] = h0 ping-pong, [2..3] = h1 ping-pong.  ~33 KB.
  __shared__ __attribute__((aligned(32))) __bf16 hlds[4][16 * LSTR];

  const int tid  = threadIdx.x;
  const int lane = tid & 31;
  const int jt   = tid >> 5;          // 0..15: this wave's hidden tile
  const int m0   = blockIdx.x << 4;   // global batch row base
  const int j0   = jt << 4;
  const int lrow = lane & 15, lhi = lane >> 4;

  // zero-init LDS h state (h_{-1} = 0)
  {
    unsigned* z = (unsigned*)hlds;
    for (int i = tid; i < 4 * 16 * LSTR / 2; i += NTHR) z[i] = 0u;
  }
  __syncthreads();

  float c0r[8], c1r[8];
#pragma unroll
  for (int v = 0; v < 8; ++v) { c0r[v] = 0.0f; c1r[v] = 0.0f; }

  for (int t = 0; t < SEQ; ++t) {
    // Defeat LICM of the (loop-invariant) weight-fragment loads: without this
    // the compiler hoists ~96 B-fragments out of the t-loop and spills them to
    // scratch ("Folded Reload"), privatizing the shared L2-resident stream.
    asm volatile("" ::: "memory");

    const __bf16* h0prev = hlds[(t + 1) & 1];
    __bf16*       h0cur  = hlds[t & 1];
    const __bf16* h1prev = hlds[2 + ((t + 1) & 1)];
    __bf16*       h1cur  = hlds[2 + (t & 1)];

    // ---------- layer 0 ----------
    {
      v8f acc[4];
#pragma unroll
      for (int v = 0; v < 8; ++v) {
        int b = m0 + (lhi << 3) + v;
        int xv = x[b * SEQ + t];
        const float* grow = G0 + xv * G4 + j0 + lrow;
        acc[0][v] = grow[0];   acc[1][v] = grow[256];
        acc[2][v] = grow[512]; acc[3][v] = grow[768];
      }
      for (int kb = 0; kb < 8; ++kb) {
        v16bf a = load_afrag_lds(h0prev, kb, lrow, lhi);
        __builtin_prefetch(W0p + ((jt * 8 + ((kb + 1) & 7)) << 9) + (lane << 4), 0, 3);
#pragma unroll
        for (int ch = 0; ch < 4; ++ch) {
          v16bf bf = *(const v16bf*)(W0p + ((((ch << 4) + jt) * 8 + kb) << 9) + (lane << 4));
          acc[ch] = __builtin_amdgcn_wmma_f32_16x16x32_bf16(
              false, a, false, bf, (short)0, acc[ch], false, false);
        }
      }
#pragma unroll
      for (int v = 0; v < 8; ++v) {
        float i_ = sigm(acc[0][v]);
        float f_ = sigm(acc[1][v]);
        float g_ = tanhf(acc[2][v]);
        float o_ = sigm(acc[3][v]);
        c0r[v] = f_ * c0r[v] + i_ * g_;
        float hv = o_ * tanhf(c0r[v]);
        h0cur[(v + (lhi << 3)) * LSTR + j0 + lrow] = f2bf(hv);
      }
    }
    __syncthreads();   // h0_t visible to all waves

    // ---------- layer 1 ----------
    {
      v8f acc[4];
#pragma unroll
      for (int ch = 0; ch < 4; ++ch) {
        float bb = b1[ch * 256 + j0 + lrow];
#pragma unroll
        for (int v = 0; v < 8; ++v) acc[ch][v] = bb;
      }
      // Wih1 . h0_t
      for (int kb = 0; kb < 8; ++kb) {
        v16bf a = load_afrag_lds(h0cur, kb, lrow, lhi);
        __builtin_prefetch(W1ip + ((jt * 8 + ((kb + 1) & 7)) << 9) + (lane << 4), 0, 3);
#pragma unroll
        for (int ch = 0; ch < 4; ++ch) {
          v16bf bf = *(const v16bf*)(W1ip + ((((ch << 4) + jt) * 8 + kb) << 9) + (lane << 4));
          acc[ch] = __builtin_amdgcn_wmma_f32_16x16x32_bf16(
              false, a, false, bf, (short)0, acc[ch], false, false);
        }
      }
      // Whh1 . h1_{t-1}
      for (int kb = 0; kb < 8; ++kb) {
        v16bf a = load_afrag_lds(h1prev, kb, lrow, lhi);
        __builtin_prefetch(W1hp + ((jt * 8 + ((kb + 1) & 7)) << 9) + (lane << 4), 0, 3);
#pragma unroll
        for (int ch = 0; ch < 4; ++ch) {
          v16bf bf = *(const v16bf*)(W1hp + ((((ch << 4) + jt) * 8 + kb) << 9) + (lane << 4));
          acc[ch] = __builtin_amdgcn_wmma_f32_16x16x32_bf16(
              false, a, false, bf, (short)0, acc[ch], false, false);
        }
      }
#pragma unroll
      for (int v = 0; v < 8; ++v) {
        float i_ = sigm(acc[0][v]);
        float f_ = sigm(acc[1][v]);
        float g_ = tanhf(acc[2][v]);
        float o_ = sigm(acc[3][v]);
        c1r[v] = f_ * c1r[v] + i_ * g_;
        float hv = o_ * tanhf(c1r[v]);
        h1cur[(v + (lhi << 3)) * LSTR + j0 + lrow] = f2bf(hv);
      }
    }
    __syncthreads();   // h1_t visible; protects next step's buffer reuse
  }

  // ----- final FC for this WG's 16 rows: out[b][v] = h1_{T-1}[b,:].Wfc[v,:]+bfc -----
  const __bf16* hf = hlds[2 + ((SEQ - 1) & 1)];
  for (int i = tid; i < 16 * VOCAB; i += NTHR) {
    int bl = i / VOCAB, vv = i - bl * VOCAB;
    float s = bfc[vv];
    const __bf16* hrow = hf + bl * LSTR;
    const float* wrow = Wfc + vv * HID;
    for (int j = 0; j < HID; ++j) s += bf2f(hrow[j]) * wrow[j];
    out[(m0 + bl) * VOCAB + vv] = s;
  }
}

// ---------------------------------------------------------------------------
extern "C" void kernel_launch(void* const* d_in, const int* in_sizes, int n_in,
                              void* d_out, int out_size, void* d_ws, size_t ws_size,
                              hipStream_t stream) {
  (void)in_sizes; (void)n_in; (void)out_size;
  const int*   x    = (const int*)d_in[0];
  const float* E    = (const float*)d_in[1];
  const float* Wih0 = (const float*)d_in[2];
  const float* Whh0 = (const float*)d_in[3];
  const float* bih0 = (const float*)d_in[4];
  const float* bhh0 = (const float*)d_in[5];
  const float* Wih1 = (const float*)d_in[6];
  const float* Whh1 = (const float*)d_in[7];
  const float* bih1 = (const float*)d_in[8];
  const float* bhh1 = (const float*)d_in[9];
  const float* Wfc  = (const float*)d_in[10];
  const float* bfc  = (const float*)d_in[11];
  float* out = (float*)d_out;

  char* ws = (char*)d_ws;
  if (ws_size < WS_NEED) return;   // needs ~1.9 MB scratch

  float*  G0   = (float*)(ws + G0_OFF);
  float*  b1   = (float*)(ws + B1_OFF);
  __bf16* W0p  = (__bf16*)(ws + W0P_OFF);
  __bf16* W1ip = (__bf16*)(ws + W1IP_OFF);
  __bf16* W1hp = (__bf16*)(ws + W1HP_OFF);

  prep_tables<<<(VOCAB * G4 + G4 + 255) / 256, 256, 0, stream>>>(
      E, Wih0, bih0, bhh0, bih1, bhh1, G0, b1);
  prep_pack<<<(3 * G4 * HID + 255) / 256, 256, 0, stream>>>(
      Whh0, Wih1, Whh1, W0p, W1ip, W1hp);
  lstm_batchtile<<<NWG, NTHR, 0, stream>>>(
      x, G0, b1, W0p, W1ip, W1hp, Wfc, bfc, out);
}
